// MLPWithAttention_88802743812924
// MI455X (gfx1250) — matrix operand order repaired and checked by reading, not compiled, not run on hardware
//
#include <hip/hip_runtime.h>
#include <hip/hip_bf16.h>

typedef __attribute__((ext_vector_type(16))) __bf16 v16bf;
typedef __attribute__((ext_vector_type(8)))  float  v8f;
typedef __attribute__((ext_vector_type(4)))  unsigned int u32x4;
typedef __attribute__((ext_vector_type(4)))  float  f32x4;
typedef unsigned short u16;

#if defined(__gfx1250__) && __has_builtin(__builtin_amdgcn_global_load_async_to_lds_b128)
#define USE_ASYNC_LDS 1
#else
#define USE_ASYNC_LDS 0
#endif

typedef int gv4i __attribute__((vector_size(16)));                 // int __vector(4)
typedef __attribute__((address_space(1))) gv4i* as1_v4;
typedef __attribute__((address_space(3))) gv4i* as3_v4;

// per-lane 16B async copy global -> LDS (ASYNCcnt-tracked on gfx1250)
__device__ __forceinline__ void cp_async16(const u16* g, u16* l) {
#if USE_ASYNC_LDS
    as1_v4 gp = (as1_v4)(unsigned long long)(const void*)g;
    as3_v4 lp = (as3_v4)(unsigned int)(unsigned long long)(void*)l;  // low 32 bits = LDS offset
    __builtin_amdgcn_global_load_async_to_lds_b128(gp, lp, 0, 0);
#else
    *(u32x4*)l = *(const u32x4*)g;
#endif
}

__device__ __forceinline__ void wait_async0() {
#if USE_ASYNC_LDS
#if __has_builtin(__builtin_amdgcn_s_wait_asynccnt)
    __builtin_amdgcn_s_wait_asynccnt(0);
#else
    asm volatile("s_wait_asynccnt 0x0" ::: "memory");
#endif
#endif
}

// ---------- bf16 helpers (bit-level, RNE) ----------
__device__ __forceinline__ float bf2f(u16 u) {
    unsigned int x = ((unsigned int)u) << 16;
    float f; __builtin_memcpy(&f, &x, 4); return f;
}
__device__ __forceinline__ u16 f2bf(float f) {
    unsigned int x; __builtin_memcpy(&x, &f, 4);
    unsigned int r = x + 0x7fffu + ((x >> 16) & 1u);   // round to nearest even
    return (u16)(r >> 16);
}

// ---------- fp32 -> bf16 conversion ----------
__global__ void f32_to_bf16_kernel(const float* __restrict__ src,
                                   u16* __restrict__ dst, long long n) {
    long long i = ((long long)blockIdx.x * blockDim.x + threadIdx.x) * 4;
    if (i + 3 < n) {
        f32x4 f = *(const f32x4*)(src + i);
        dst[i + 0] = f2bf(f.x); dst[i + 1] = f2bf(f.y);
        dst[i + 2] = f2bf(f.z); dst[i + 3] = f2bf(f.w);
    } else {
        for (long long j = i; j < n; ++j) dst[j] = f2bf(src[j]);
    }
}

// ---------- tiled bf16 transpose: (rows x cols) -> (cols x rows), batched ----------
__global__ void transpose_bf16_kernel(const u16* __restrict__ in, u16* __restrict__ out,
                                      int rows, int cols) {
    __shared__ u16 t[32][33];
    long long boff = (long long)blockIdx.z * rows * cols;
    int c0 = blockIdx.x * 32, r0 = blockIdx.y * 32;
    for (int i = threadIdx.y; i < 32; i += 8)
        t[i][threadIdx.x] = in[boff + (long long)(r0 + i) * cols + (c0 + threadIdx.x)];
    __syncthreads();
    for (int i = threadIdx.y; i < 32; i += 8)
        out[boff + (long long)(c0 + i) * rows + (r0 + threadIdx.x)] = t[threadIdx.x][i];
}

// ---------- row softmax: fp32 in, bf16 out ----------
__global__ void softmax_rows_kernel(const float* __restrict__ S, u16* __restrict__ O, int cols) {
    __shared__ float red[256];
    long long row = blockIdx.x;
    const float* p = S + row * (long long)cols;
    u16* o = O + row * (long long)cols;
    float m = -3.4e38f;
    for (int c = threadIdx.x; c < cols; c += 256) m = fmaxf(m, p[c]);
    red[threadIdx.x] = m; __syncthreads();
    for (int s = 128; s > 0; s >>= 1) {
        if ((int)threadIdx.x < s) red[threadIdx.x] = fmaxf(red[threadIdx.x], red[threadIdx.x + s]);
        __syncthreads();
    }
    m = red[0]; __syncthreads();
    float sum = 0.f;
    for (int c = threadIdx.x; c < cols; c += 256) sum += __expf(p[c] - m);
    red[threadIdx.x] = sum; __syncthreads();
    for (int s = 128; s > 0; s >>= 1) {
        if ((int)threadIdx.x < s) red[threadIdx.x] += red[threadIdx.x + s];
        __syncthreads();
    }
    float inv = 1.f / red[0];
    for (int c = threadIdx.x; c < cols; c += 256) o[c] = f2bf(__expf(p[c] - m) * inv);
}

// ---------- WMMA GEMM: C[m,n] = act(scale * sum_k A[m,k]*W[n,k] + bias[n] + resid[m,n]) ----------
// A: (M,K) bf16 row-major. W: (N,K) bf16 row-major. Block tile 128(M)x256(N), 8 waves in 2Mx4N,
// wave tile 64x64 = 4x4 v_wmma_f32_16x16x32_bf16. LDS double-buffered, async global->LDS copies
// of tile k+1 overlap WMMA of tile k. Batched via grid.z strides.
union FragU { u32x4 q[2]; v16bf v; };

__global__ __launch_bounds__(256)
void gemm_wmma_nt(const u16* __restrict__ A, long long sA,
                  const u16* __restrict__ W, long long sW,
                  const float* __restrict__ bias,
                  const u16* __restrict__ resid, long long sR,
                  u16* __restrict__ outB, long long sOB,
                  float* __restrict__ outF, long long sOF,
                  int M, int N, int K, float scale, int relu) {
    __shared__ u16 As[2][128 * 32];   // 2 x 8 KB
    __shared__ u16 Bs[2][256 * 32];   // 2 x 16 KB

    int bz = blockIdx.z;
    A += (long long)bz * sA;
    W += (long long)bz * sW;
    if (resid) resid += (long long)bz * sR;
    if (outB)  outB  += (long long)bz * sOB;
    if (outF)  outF  += (long long)bz * sOF;

    const int tile_m = blockIdx.y * 128;
    const int tile_n = blockIdx.x * 256;

    const int tid  = threadIdx.x;
    const int lane = tid & 31;
    const int wave = tid >> 5;
    const int wm   = (wave & 1) * 64;    // wave M offset in tile
    const int wn   = (wave >> 1) * 64;   // wave N offset in tile
    const int lr   = lane & 15;
    const bool hi  = lane >= 16;

    // staging map: A tile 128x32 -> 2 thr/row x 32B; B tile 256x32 -> 1 thr/row x 64B
    const int arow = tid >> 1;
    const int acol = (tid & 1) * 16;
    const int brow = tid;

    auto stage = [&](int k0, int buf) {
        const u16* gA = A + (long long)(tile_m + arow) * K + k0 + acol;
        u16*       lA = &As[buf][arow * 32 + acol];
        const u16* gB = W + (long long)(tile_n + brow) * K + k0;
        u16*       lB = &Bs[buf][brow * 32];
        cp_async16(gA,      lA);
        cp_async16(gA + 8,  lA + 8);
        cp_async16(gB,      lB);
        cp_async16(gB + 8,  lB + 8);
        cp_async16(gB + 16, lB + 16);
        cp_async16(gB + 24, lB + 24);
    };

    v8f acc[4][4];
    const v8f vz = {0.f, 0.f, 0.f, 0.f, 0.f, 0.f, 0.f, 0.f};
#pragma unroll
    for (int i = 0; i < 4; ++i)
#pragma unroll
        for (int j = 0; j < 4; ++j) acc[i][j] = vz;

    const int nk = K >> 5;
    stage(0, 0);

    for (int kb = 0; kb < nk; ++kb) {
        wait_async0();
        __syncthreads();                       // tile kb resident, all readers of buf^1 done
        if (kb + 1 < nk) stage((kb + 1) << 5, (kb + 1) & 1);   // DMA next tile during compute
        const int buf = kb & 1;

        // B fragments: lane L -> column n0 + (L%16), 16 contiguous K at (hi?16:0)
        v16bf bfr[4];
#pragma unroll
        for (int j = 0; j < 4; ++j) {
            const u16* p = &Bs[buf][(wn + j * 16 + lr) * 32 + (hi ? 16 : 0)];
            FragU f; f.q[0] = *(const u32x4*)p; f.q[1] = *(const u32x4*)(p + 8);
            bfr[j] = f.v;
        }
        // A fragments: lane L -> row m0 + (L%16), 16 contiguous K at (hi?16:0)
#pragma unroll
        for (int i = 0; i < 4; ++i) {
            const u16* p = &As[buf][(wm + i * 16 + lr) * 32 + (hi ? 16 : 0)];
            FragU f; f.q[0] = *(const u32x4*)p; f.q[1] = *(const u32x4*)(p + 8);
            v16bf af = f.v;
#pragma unroll
            for (int j = 0; j < 4; ++j)
                acc[i][j] = __builtin_amdgcn_wmma_f32_16x16x32_bf16(
                    false, af, false, bfr[j], (short)0, acc[i][j], false, false);
        }
    }

    // epilogue: C f32 layout: VGPR r -> row m0 + (hi?8:0) + r, col n0 + (lane%16)
#pragma unroll
    for (int i = 0; i < 4; ++i) {
#pragma unroll
        for (int j = 0; j < 4; ++j) {
            const int gn = tile_n + wn + j * 16 + lr;
            const float bv = bias ? bias[gn] : 0.f;
#pragma unroll
            for (int r = 0; r < 8; ++r) {
                const int gm = tile_m + wm + i * 16 + (hi ? 8 : 0) + r;
                float val = acc[i][j][r] * scale + bv;
                const long long idx = (long long)gm * N + gn;
                if (resid) val += bf2f(resid[idx]);
                if (relu)  val = fmaxf(val, 0.f);
                if (outB) outB[idx] = f2bf(val);
                if (outF) outF[idx] = val;
            }
        }
    }
}

// ---------- host side ----------
static inline void launch_gemm(hipStream_t s,
                               const u16* A, long long sA, const u16* W, long long sW,
                               const float* bias, const u16* resid, long long sR,
                               u16* outB, long long sOB, float* outF, long long sOF,
                               int M, int N, int K, float scale, int relu, int batch) {
    dim3 grid(N / 256, M / 128, batch), block(256);
    hipLaunchKernelGGL(gemm_wmma_nt, grid, block, 0, s,
                       A, sA, W, sW, bias, resid, sR, outB, sOB, outF, sOF,
                       M, N, K, scale, relu);
}

extern "C" void kernel_launch(void* const* d_in, const int* in_sizes, int n_in,
                              void* d_out, int out_size, void* d_ws, size_t ws_size,
                              hipStream_t stream) {
    const int B = 4, S = 2048, DIN = 1024, H1 = 2048, H2 = 1024, DOUT = 1024;
    const long long BS = (long long)B * S;                 // 8192 rows

    const float* x  = (const float*)d_in[0];
    const float* w1 = (const float*)d_in[1];  const float* b1 = (const float*)d_in[2];
    const float* w2 = (const float*)d_in[3];  const float* b2 = (const float*)d_in[4];
    const float* wq = (const float*)d_in[5];  const float* bq = (const float*)d_in[6];
    const float* wk = (const float*)d_in[7];  const float* bk = (const float*)d_in[8];
    const float* wv = (const float*)d_in[9];  const float* bv = (const float*)d_in[10];
    const float* wo = (const float*)d_in[11]; const float* bo = (const float*)d_in[12];
    float* out = (float*)d_out;

    // workspace carve-out (256B aligned)
    char* ws = (char*)d_ws;
    size_t off = 0;
    auto carve = [&](size_t bytes) { size_t o = off; off = (off + bytes + 255) & ~(size_t)255; return o; };
    u16* xb   = (u16*)(ws + carve(BS * DIN * 2));
    u16* w1b  = (u16*)(ws + carve((size_t)H1 * DIN * 2));
    u16* w2b  = (u16*)(ws + carve((size_t)H2 * H1 * 2));
    u16* wqb  = (u16*)(ws + carve((size_t)H2 * H2 * 2));
    u16* wkb  = (u16*)(ws + carve((size_t)H2 * H2 * 2));
    u16* wvb  = (u16*)(ws + carve((size_t)H2 * H2 * 2));
    u16* wob  = (u16*)(ws + carve((size_t)DOUT * H2 * 2));
    u16* h1   = (u16*)(ws + carve(BS * H1 * 2));
    u16* h2   = (u16*)(ws + carve(BS * H2 * 2));
    u16* qb   = (u16*)(ws + carve(BS * H2 * 2));
    u16* kb   = (u16*)(ws + carve(BS * H2 * 2));
    u16* vb   = (u16*)(ws + carve(BS * H2 * 2));
    u16* vtb  = (u16*)(ws + carve(BS * H2 * 2));                  // V^T per batch (H2 x S)
    float* sc = (float*)(ws + carve(BS * (size_t)S * 4));          // scores fp32
    u16* attn = (u16*)(ws + carve(BS * (size_t)S * 2));            // softmax bf16
    u16* ctx  = (u16*)(ws + carve(BS * H2 * 2));

    auto cv = [&](const float* sptr, u16* dptr, long long n) {
        long long nv = (n + 3) / 4;
        dim3 g((unsigned)((nv + 255) / 256)), blk(256);
        hipLaunchKernelGGL(f32_to_bf16_kernel, g, blk, 0, stream, sptr, dptr, n);
    };
    cv(x,  xb,  BS * DIN);
    cv(w1, w1b, (long long)H1 * DIN);
    cv(w2, w2b, (long long)H2 * H1);
    cv(wq, wqb, (long long)H2 * H2);
    cv(wk, wkb, (long long)H2 * H2);
    cv(wv, wvb, (long long)H2 * H2);
    cv(wo, wob, (long long)DOUT * H2);

    // h1 = relu(x @ w1^T + b1)    (M=8192, N=2048, K=1024)
    launch_gemm(stream, xb, 0, w1b, 0, b1, nullptr, 0, h1, 0, nullptr, 0,
                (int)BS, H1, DIN, 1.f, 1, 1);
    // h2 = relu(h1 @ w2^T + b2)   (M=8192, N=1024, K=2048)
    launch_gemm(stream, h1, 0, w2b, 0, b2, nullptr, 0, h2, 0, nullptr, 0,
                (int)BS, H2, H1, 1.f, 1, 1);
    // q/k/v = h2 @ w{q,k,v}^T + b (M=8192, N=1024, K=1024)
    launch_gemm(stream, h2, 0, wqb, 0, bq, nullptr, 0, qb, 0, nullptr, 0, (int)BS, H2, H2, 1.f, 0, 1);
    launch_gemm(stream, h2, 0, wkb, 0, bk, nullptr, 0, kb, 0, nullptr, 0, (int)BS, H2, H2, 1.f, 0, 1);
    launch_gemm(stream, h2, 0, wvb, 0, bv, nullptr, 0, vb, 0, nullptr, 0, (int)BS, H2, H2, 1.f, 0, 1);

    // scores = (q @ k^T) / sqrt(H2), per batch (M=N=2048, K=1024), fp32 out
    launch_gemm(stream, qb, (long long)S * H2, kb, (long long)S * H2, nullptr, nullptr, 0,
                nullptr, 0, sc, (long long)S * S, S, S, H2, 1.f / 32.f, 0, B);

    // softmax rows (8192 rows x 2048), bf16 out
    hipLaunchKernelGGL(softmax_rows_kernel, dim3((unsigned)BS), dim3(256), 0, stream, sc, attn, S);

    // V^T per batch: (S x H2) -> (H2 x S)
    hipLaunchKernelGGL(transpose_bf16_kernel, dim3(H2 / 32, S / 32, B), dim3(32, 8), 0, stream,
                       vb, vtb, S, H2);

    // ctx = attn @ V + V  == attn(M=2048,K=2048) @ (V^T as W (N=1024,K=2048))^T + resid V
    launch_gemm(stream, attn, (long long)S * S, vtb, (long long)H2 * S, nullptr,
                vb, (long long)S * H2, ctx, (long long)S * H2, nullptr, 0,
                S, H2, S, 1.f, 0, B);

    // out = ctx @ wo^T + bo (M=8192, N=1024, K=1024), fp32 -> d_out
    launch_gemm(stream, ctx, 0, wob, 0, bo, nullptr, 0, nullptr, 0, out, 0,
                (int)BS, DOUT, H2, 1.f, 0, 1);
}
